// CorssCasualAttentionBlock_48790828482897
// MI455X (gfx1250) — matrix operand order repaired and checked by reading, not compiled, not run on hardware
//
#include <hip/hip_runtime.h>
#include <math.h>

// ---------------------------------------------------------------------------
// CDNA5 / gfx1250 transformer block:  i,c,j -> LN -> QKV -> flash block-causal
// attention (WMMA) -> Wo + residual -> LN2 -> Wfc+GELU -> Wpr + residual
// GEMMs + attention use v_wmma_f32_16x16x32_f16 (wave32).  GEMM A-tiles are
// streamed HBM->LDS by the Tensor Data Mover in a double-buffered pipeline
// (tensor_load_to_lds overlapped with WMMA, s_wait_tensorcnt fencing).
// ---------------------------------------------------------------------------

typedef __attribute__((ext_vector_type(16))) _Float16 v16h;
typedef __attribute__((ext_vector_type(8)))  _Float16 v8h;
typedef __attribute__((ext_vector_type(2)))  _Float16 v2h;
typedef __attribute__((ext_vector_type(8)))  float    v8f;
typedef __attribute__((ext_vector_type(4)))  unsigned v4u;
typedef __attribute__((ext_vector_type(8)))  int      v8i;
typedef __attribute__((ext_vector_type(4)))  int      v4i;

#define NB 8      // batch
#define NT 1024   // tokens per batch  (V*H*W = 4*256)
#define NC 512    // channels
#define NHD 8     // heads
#define HD 64     // head dim
#define HW 256    // H*W (view block size)

#if __HIP_DEVICE_COMPILE__ && __has_builtin(__builtin_amdgcn_tensor_load_to_lds)
#define USE_TDM 1
#else
#define USE_TDM 0
#endif

__device__ __forceinline__ v8f wmma16x16x32(v16h a, v16h b, v8f c) {
  return __builtin_amdgcn_wmma_f32_16x16x32_f16(false, a, false, b, (short)0, c,
                                                false, false);
}

// A fragment (16x32 f16, MxK) from a row-major LDS tile (stride in halves).
// i-th half: K = i + 16*(lane>>4), M = lane&15
__device__ __forceinline__ v16h load_a_frag(const _Float16* tile, int stride,
                                            int lmod, int lhalf) {
  v16h a;
  const _Float16* row = tile + lmod * stride;
#pragma unroll
  for (int p = 0; p < 8; ++p) {
    int k = 2 * p + 16 * lhalf;
    v2h t = *(const v2h*)(row + k);
    a[2 * p] = t[0];
    a[2 * p + 1] = t[1];
  }
  return a;
}

// B fragment (32x16 f16, KxN) from an N-major (transposed) LDS tile:
// tile[n][k], i-th half: K = i + 16*(lane>>4), N = lane&15
__device__ __forceinline__ v16h load_b_frag(const _Float16* tileT, int stride,
                                            int lmod, int lhalf) {
  v16h b;
  const _Float16* col = tileT + lmod * stride + 16 * lhalf;
#pragma unroll
  for (int p = 0; p < 8; ++p) {
    v2h t = *(const v2h*)(col + 2 * p);
    b[2 * p] = t[0];
    b[2 * p + 1] = t[1];
  }
  return b;
}

__device__ __forceinline__ float gelu_new(float x) {
  float x3 = x * x * x;
  return 0.5f * x * (1.0f + tanhf(0.7978845608028654f * (x + 0.044715f * x3)));
}

#if USE_TDM
// Issue a TDM load of a 32(K) x 128(M) f16 tile of A[M,K] into LDS with a
// 36-half padded row stride (pad_interval = 16 DWORDs, pad_amount = 2 DWORDs).
__device__ __forceinline__ void tdm_load_a_tile(unsigned lds_off,
                                                const _Float16* gsrc, int M,
                                                int K) {
  unsigned long long ga = (unsigned long long)(uintptr_t)gsrc;
  v4u g0;
  g0[0] = 1u;                                   // count=1 (user D#)
  g0[1] = lds_off;                              // lds_addr
  g0[2] = (unsigned)(ga & 0xffffffffu);         // global_addr lo
  g0[3] = (unsigned)((ga >> 32) & 0x01ffffffu)  // global_addr hi
          | (2u << 30);                         // type=2 (image)
  v8i g1;
  // data_size=2B, pad_enable, pad_interval=16DW(one row), pad_amount=2DW
  g1[0] = (int)((1u << 16) | (1u << 20) | (3u << 22) | (1u << 25));
  g1[1] = (int)((unsigned)(K & 0xffff) << 16);              // tensor_dim0 lo
  g1[2] = (int)(((unsigned)K >> 16) | ((unsigned)(M & 0xffff) << 16));
  g1[3] = (int)(((unsigned)M >> 16) | (32u << 16));         // tile_dim0=32
  g1[4] = 128;                                              // tile_dim1=128
  g1[5] = K;                                                // dim0 stride
  g1[6] = 0;
  g1[7] = 0;
  v4i gz = {0, 0, 0, 0};
#if __clang_major__ >= 23
  v8i gz8 = {0, 0, 0, 0, 0, 0, 0, 0};
  __builtin_amdgcn_tensor_load_to_lds(g0, g1, gz, gz, gz8, 0);
#else
  __builtin_amdgcn_tensor_load_to_lds(g0, g1, gz, gz, 0);
#endif
}
#else
__device__ __forceinline__ void stage_a_manual(_Float16* dst,
                                               const _Float16* A, int bm,
                                               int K, int k0, int tid) {
#pragma unroll
  for (int it = 0; it < 16; ++it) {
    int idx = tid + it * 128;  // 2048 half2
    int r = idx >> 4, c = (idx & 15) * 2;
    *(v2h*)(dst + r * 36 + c) = *(const v2h*)(A + (size_t)(bm + r) * K + k0 + c);
  }
}
#endif

// ---------------------------------------------------------------------------
__global__ void cvt_f32_f16_kernel(const float* __restrict__ src,
                                   _Float16* __restrict__ dst, int n) {
  int i = blockIdx.x * blockDim.x + threadIdx.x;
  for (; i < n; i += gridDim.x * blockDim.x) dst[i] = (_Float16)src[i];
}

// ---------------------------------------------------------------------------
__device__ __forceinline__ float block_allreduce_sum256(float v) {
  __shared__ float red[8];
#pragma unroll
  for (int off = 16; off > 0; off >>= 1) v += __shfl_xor(v, off, 32);
  if ((threadIdx.x & 31) == 0) red[threadIdx.x >> 5] = v;
  __syncthreads();
  float s = 0.f;
#pragma unroll
  for (int i = 0; i < 8; ++i) s += red[i];
  __syncthreads();
  return s;
}

// ---------------------------------------------------------------------------
// (v,b,c,h,w) -> (b,t,c) gather + ln1 for i, j, c
// ---------------------------------------------------------------------------
__global__ __launch_bounds__(256) void ln_in_kernel(
    const float* __restrict__ iS, const float* __restrict__ cS,
    const float* __restrict__ jS, const float* __restrict__ g,
    const float* __restrict__ bt, _Float16* __restrict__ qs,
    _Float16* __restrict__ kvs, float* __restrict__ c_ln) {
  int row = blockIdx.x;  // b*NT + t
  int b = row >> 10, t = row & (NT - 1);
  int v = t >> 8, hw = t & (HW - 1);
  size_t base = ((size_t)(v * NB + b) * NC) * HW + hw;  // + ch*HW
  int c0 = threadIdx.x, c1 = threadIdx.x + 256;

  const float* srcs[3] = {iS, jS, cS};
#pragma unroll 1
  for (int s = 0; s < 3; ++s) {
    const float* x = srcs[s];
    float x0 = x[base + (size_t)c0 * HW];
    float x1 = x[base + (size_t)c1 * HW];
    float mu = block_allreduce_sum256(x0 + x1) * (1.0f / NC);
    float d0 = x0 - mu, d1 = x1 - mu;
    float var = block_allreduce_sum256(d0 * d0 + d1 * d1) * (1.0f / NC);
    float rstd = rsqrtf(var + 1e-5f);
    float y0 = d0 * rstd * g[c0] + bt[c0];
    float y1 = d1 * rstd * g[c1] + bt[c1];
    size_t o0 = (size_t)row * NC + c0, o1 = (size_t)row * NC + c1;
    if (s == 0) {
      qs[o0] = (_Float16)y0; qs[o1] = (_Float16)y1;
    } else if (s == 1) {
      kvs[o0] = (_Float16)y0; kvs[o1] = (_Float16)y1;
    } else {
      c_ln[o0] = y0; c_ln[o1] = y1;
    }
  }
}

// ---------------------------------------------------------------------------
__global__ __launch_bounds__(256) void ln_row_kernel(
    const float* __restrict__ xin, const float* __restrict__ g,
    const float* __restrict__ bt, _Float16* __restrict__ out) {
  int row = blockIdx.x;
  const float* x = xin + (size_t)row * NC;
  int c0 = threadIdx.x, c1 = threadIdx.x + 256;
  float x0 = x[c0], x1 = x[c1];
  float mu = block_allreduce_sum256(x0 + x1) * (1.0f / NC);
  float d0 = x0 - mu, d1 = x1 - mu;
  float var = block_allreduce_sum256(d0 * d0 + d1 * d1) * (1.0f / NC);
  float rstd = rsqrtf(var + 1e-5f);
  out[(size_t)row * NC + c0] = (_Float16)(d0 * rstd * g[c0] + bt[c0]);
  out[(size_t)row * NC + c1] = (_Float16)(d1 * rstd * g[c1] + bt[c1]);
}

// ---------------------------------------------------------------------------
// Generic WMMA GEMM:  C[M,N] = act(A[M,K] @ W[K,N] + bias) (+residual)
// 128 threads = 4 waves; block tile 128x64, wave tile 32x64, BK=32.
// Double-buffered pipeline: TDM streams A tile k+1 while WMMAs consume tile
// k; W tile k+1 is pipelined through registers into the transposed LDS tile.
// EPI 0: f16 out *scale | 1: f32 out + resid | 2: gelu->f16 | 3: scatter+resid
// ---------------------------------------------------------------------------
template <int EPI>
__global__ __launch_bounds__(128) void gemm_wmma_kernel(
    const _Float16* __restrict__ A, const _Float16* __restrict__ W,
    const float* __restrict__ bias, _Float16* __restrict__ outH,
    float* __restrict__ outF, const float* __restrict__ resid, int M, int N,
    int K, float scaleOut) {
  __shared__ _Float16 sA[2][128 * 36];
  __shared__ _Float16 sB[2][64 * 36];  // transposed: sB[.][n*36 + k]

  const int bm = blockIdx.x * 128, bn = blockIdx.y * 64;
  const int wave = threadIdx.x >> 5, lane = threadIdx.x & 31;
  const int lmod = lane & 15, lhalf = lane >> 4;

  v8f acc[2][4] = {};

  // ---- prologue: stage tile k0 = 0 into buffer 0 ----
#if USE_TDM
  const unsigned sAoff[2] = {(unsigned)(uintptr_t)(&sA[0][0]),
                             (unsigned)(uintptr_t)(&sA[1][0])};
  if (wave == 0) tdm_load_a_tile(sAoff[0], A + (size_t)bm * K, M, K);
#else
  stage_a_manual(&sA[0][0], A, bm, K, 0, threadIdx.x);
#endif
  {
#pragma unroll
    for (int it = 0; it < 8; ++it) {
      int idx = threadIdx.x + it * 128;  // 1024 half2
      int r = idx >> 5, c = (idx & 31) * 2;
      v2h t = *(const v2h*)(W + (size_t)r * N + bn + c);
      sB[0][(c + 0) * 36 + r] = t[0];
      sB[0][(c + 1) * 36 + r] = t[1];
    }
  }
#if USE_TDM
  if (wave == 0) __builtin_amdgcn_s_wait_tensorcnt(0);
#endif
  __syncthreads();

  // ---- main pipelined loop ----
  int cur = 0;
  for (int k0 = 0; k0 < K; k0 += 32) {
    const int kn = k0 + 32;
    const bool more = kn < K;

#if USE_TDM
    if (more && wave == 0)  // stream next A tile while we compute this one
      tdm_load_a_tile(sAoff[cur ^ 1], A + (size_t)bm * K + kn, M, K);
#endif
    v2h wreg[8];
    if (more) {  // next W tile -> registers (overlaps compute)
#pragma unroll
      for (int it = 0; it < 8; ++it) {
        int idx = threadIdx.x + it * 128;
        int r = idx >> 5, c = (idx & 31) * 2;
        wreg[it] = *(const v2h*)(W + (size_t)(kn + r) * N + bn + c);
      }
    }

    // ---- compute from current buffers: 8 WMMAs ----
    v16h af0 = load_a_frag(&sA[cur][(wave * 32 + 0) * 36], 36, lmod, lhalf);
    v16h af1 = load_a_frag(&sA[cur][(wave * 32 + 16) * 36], 36, lmod, lhalf);
#pragma unroll
    for (int nt = 0; nt < 4; ++nt) {
      v16h bf = load_b_frag(&sB[cur][nt * 16 * 36], 36, lmod, lhalf);
      acc[0][nt] = wmma16x16x32(af0, bf, acc[0][nt]);
      acc[1][nt] = wmma16x16x32(af1, bf, acc[1][nt]);
    }
    __syncthreads();  // everyone done reading buffers `cur`

    if (more) {
      _Float16* dst = &sB[cur ^ 1][0];
#pragma unroll
      for (int it = 0; it < 8; ++it) {
        int idx = threadIdx.x + it * 128;
        int r = idx >> 5, c = (idx & 31) * 2;
        dst[(c + 0) * 36 + r] = wreg[it][0];
        dst[(c + 1) * 36 + r] = wreg[it][1];
      }
#if USE_TDM
      if (wave == 0) __builtin_amdgcn_s_wait_tensorcnt(0);
#else
      stage_a_manual(&sA[cur ^ 1][0], A, bm, K, kn, threadIdx.x);
#endif
      __syncthreads();  // publish buffers `cur^1`
      cur ^= 1;
    }
  }

  // ---- epilogue: element (M = mi*16 + r + 8*lhalf, N = nt*16 + lmod) ----
#pragma unroll
  for (int mi = 0; mi < 2; ++mi)
#pragma unroll
    for (int nt = 0; nt < 4; ++nt) {
      int n = bn + nt * 16 + lmod;
      float bv = bias[n];
#pragma unroll
      for (int r = 0; r < 8; ++r) {
        int m = bm + wave * 32 + mi * 16 + r + 8 * lhalf;
        float x = acc[mi][nt][r] + bv;
        if (EPI == 0) {
          outH[(size_t)m * N + n] = (_Float16)(x * scaleOut);
        } else if (EPI == 1) {
          outF[(size_t)m * N + n] = x + resid[(size_t)m * N + n];
        } else if (EPI == 2) {
          outH[(size_t)m * N + n] = (_Float16)gelu_new(x);
        } else {
          float val = x + resid[(size_t)m * N + n];
          int b = m >> 10, t = m & (NT - 1);
          int v = t >> 8, hw = t & (HW - 1);
          outF[(((size_t)(v * NB + b) * NC) + n) * HW + hw] = val;
        }
      }
    }
}

// ---------------------------------------------------------------------------
// Flash block-causal attention.  One wave = one 16-row query tile of one
// (b, head).  q pre-scaled by 1/sqrt(hd).  kv layout [B*T, 1024]: K cols
// 0..511, V cols 512..1023.  Mask handled by the key-loop bound.
// ---------------------------------------------------------------------------
__global__ __launch_bounds__(128) void attn_kernel(
    const _Float16* __restrict__ q, const _Float16* __restrict__ kv,
    _Float16* __restrict__ y) {
  __shared__ _Float16 pbuf[4][16 * 34];  // P re-layout (C->A) scratch
  __shared__ _Float16 vbuf[4][32 * 72];  // V chunk, row-major, 16B-aligned rows

  const int wave = threadIdx.x >> 5, lane = threadIdx.x & 31;
  const int lmod = lane & 15, lhalf = lane >> 4;
  const int tile = blockIdx.x * 4 + wave;  // [b][h][qt]
  const int qt = tile & 63;
  const int h = (tile >> 6) & 7;
  const int b = tile >> 9;
  const int q0 = qt * 16;

  // Q A-fragments (d 0..31, d 32..63)
  v16h qa[2];
  {
    const _Float16* qrow = q + ((size_t)b * NT + q0 + lmod) * NC + h * HD;
#pragma unroll
    for (int f = 0; f < 2; ++f)
#pragma unroll
      for (int p = 0; p < 8; ++p) {
        int k = 2 * p + 16 * lhalf + f * 32;
        v2h t = *(const v2h*)(qrow + k);
        qa[f][2 * p] = t[0];
        qa[f][2 * p + 1] = t[1];
      }
  }

  v8f o[4] = {};
  float mrow[8], lrow[8];
#pragma unroll
  for (int r = 0; r < 8; ++r) { mrow[r] = -1e30f; lrow[r] = 0.f; }

  const int kend = ((q0 >> 8) + 1) << 8;  // block-causal bound
  for (int k0 = 0; k0 < kend; k0 += 32) {
    // ---- stage V chunk: lane -> one 64-half token row (b128 loads) ----
    {
      const _Float16* vrow =
          kv + ((size_t)b * NT + k0 + lane) * (2 * NC) + NC + h * HD;
      _Float16* dst = &vbuf[wave][lane * 72];
#pragma unroll
      for (int u = 0; u < 8; ++u)
        *(v8h*)(dst + u * 8) = *(const v8h*)(vrow + u * 8);
    }

    // ---- scores S = Q K^T (two 16x16 tiles covering 32 keys) ----
    v8f s[2];
#pragma unroll
    for (int nt = 0; nt < 2; ++nt) {
      const _Float16* krow =
          kv + ((size_t)b * NT + k0 + nt * 16 + lmod) * (2 * NC) + h * HD;
      v16h kb0, kb1;
#pragma unroll
      for (int p = 0; p < 8; ++p) {
        v2h t0 = *(const v2h*)(krow + 16 * lhalf + 2 * p);
        kb0[2 * p] = t0[0]; kb0[2 * p + 1] = t0[1];
        v2h t1 = *(const v2h*)(krow + 32 + 16 * lhalf + 2 * p);
        kb1[2 * p] = t1[0]; kb1[2 * p + 1] = t1[1];
      }
      v8f z = {};
      z = wmma16x16x32(qa[0], kb0, z);
      s[nt] = wmma16x16x32(qa[1], kb1, z);
    }

    // ---- online softmax (rows spread over 16-lane half-groups) ----
#pragma unroll
    for (int r = 0; r < 8; ++r) {
      float mx = fmaxf(s[0][r], s[1][r]);
#pragma unroll
      for (int off = 1; off < 16; off <<= 1)
        mx = fmaxf(mx, __shfl_xor(mx, off, 32));
      float mnew = fmaxf(mrow[r], mx);
      float p0 = __expf(s[0][r] - mnew);
      float p1 = __expf(s[1][r] - mnew);
      s[0][r] = p0; s[1][r] = p1;
      float rs = p0 + p1;
#pragma unroll
      for (int off = 1; off < 16; off <<= 1) rs += __shfl_xor(rs, off, 32);
      float alpha = __expf(mrow[r] - mnew);
      lrow[r] = lrow[r] * alpha + rs;
      mrow[r] = mnew;
#pragma unroll
      for (int dt = 0; dt < 4; ++dt) o[dt][r] *= alpha;
    }

    // ---- re-layout P (C-layout -> A-layout) through per-wave LDS ----
    _Float16* pb = &pbuf[wave][0];
#pragma unroll
    for (int nt = 0; nt < 2; ++nt)
#pragma unroll
      for (int r = 0; r < 8; ++r)
        pb[(r + 8 * lhalf) * 34 + nt * 16 + lmod] = (_Float16)s[nt][r];
    asm volatile("s_wait_dscnt 0" ::: "memory");

    v16h pa;
    {
      const _Float16* prow = pb + lmod * 34;
#pragma unroll
      for (int p = 0; p < 8; ++p) {
        int k = 2 * p + 16 * lhalf;
        v2h t = *(const v2h*)(prow + k);
        pa[2 * p] = t[0];
        pa[2 * p + 1] = t[1];
      }
    }

    // ---- O += P V  (V B-frags gathered from the LDS-staged chunk) ----
#pragma unroll
    for (int dt = 0; dt < 4; ++dt) {
      v16h vb;
#pragma unroll
      for (int i2 = 0; i2 < 16; ++i2)
        vb[i2] = vbuf[wave][(i2 + 16 * lhalf) * 72 + dt * 16 + lmod];
      o[dt] = wmma16x16x32(pa, vb, o[dt]);
    }
  }

  // ---- normalize + store y (f16, [B*T, C]) ----
#pragma unroll
  for (int dt = 0; dt < 4; ++dt)
#pragma unroll
    for (int r = 0; r < 8; ++r) {
      int m = q0 + r + 8 * lhalf;
      float val = o[dt][r] / lrow[r];
      y[((size_t)b * NT + m) * NC + h * HD + dt * 16 + lmod] = (_Float16)val;
    }
}

// ---------------------------------------------------------------------------
extern "C" void kernel_launch(void* const* d_in, const int* in_sizes, int n_in,
                              void* d_out, int out_size, void* d_ws,
                              size_t ws_size, hipStream_t stream) {
  const float* iS  = (const float*)d_in[0];
  const float* cS  = (const float*)d_in[1];
  const float* jS  = (const float*)d_in[2];
  const float* ln1g = (const float*)d_in[3];
  const float* ln1b = (const float*)d_in[4];
  const float* ln2g = (const float*)d_in[5];
  const float* ln2b = (const float*)d_in[6];
  const float* Wq  = (const float*)d_in[7];
  const float* bq  = (const float*)d_in[8];
  const float* Wkv = (const float*)d_in[9];
  const float* bkv = (const float*)d_in[10];
  const float* Wo  = (const float*)d_in[11];
  const float* bo  = (const float*)d_in[12];
  const float* Wfc = (const float*)d_in[13];
  const float* bfc = (const float*)d_in[14];
  const float* Wpr = (const float*)d_in[15];
  const float* bpr = (const float*)d_in[16];
  float* out = (float*)d_out;

  const int M = NB * NT;  // 8192 token rows
  char* ws = (char*)d_ws;
  size_t off = 0;
  auto alloc = [&](size_t bytes) -> char* {
    char* p = ws + off;
    off += (bytes + 255) & ~(size_t)255;
    return p;
  };

  _Float16* WqH  = (_Float16*)alloc((size_t)NC * NC * 2);
  _Float16* WkvH = (_Float16*)alloc((size_t)NC * 2 * NC * 2);
  _Float16* WoH  = (_Float16*)alloc((size_t)NC * NC * 2);
  _Float16* WfcH = (_Float16*)alloc((size_t)NC * 4 * NC * 2);
  _Float16* WprH = (_Float16*)alloc((size_t)4 * NC * NC * 2);
  _Float16* qsH  = (_Float16*)alloc((size_t)M * NC * 2);
  _Float16* kvsH = (_Float16*)alloc((size_t)M * NC * 2);
  _Float16* qH   = (_Float16*)alloc((size_t)M * NC * 2);
  _Float16* kvH  = (_Float16*)alloc((size_t)M * 2 * NC * 2);
  _Float16* yH   = (_Float16*)alloc((size_t)M * NC * 2);
  float*    cLn  = (float*)alloc((size_t)M * NC * 4);
  float*    cs   = (float*)alloc((size_t)M * NC * 4);
  _Float16* cs2H = (_Float16*)alloc((size_t)M * NC * 2);
  _Float16* h1H  = (_Float16*)alloc((size_t)M * 4 * NC * 2);
  (void)ws_size; (void)in_sizes; (void)n_in; (void)out_size;

  // 1) weights fp32 -> fp16
  cvt_f32_f16_kernel<<<512, 256, 0, stream>>>(Wq, WqH, NC * NC);
  cvt_f32_f16_kernel<<<512, 256, 0, stream>>>(Wkv, WkvH, NC * 2 * NC);
  cvt_f32_f16_kernel<<<512, 256, 0, stream>>>(Wo, WoH, NC * NC);
  cvt_f32_f16_kernel<<<512, 256, 0, stream>>>(Wfc, WfcH, NC * 4 * NC);
  cvt_f32_f16_kernel<<<512, 256, 0, stream>>>(Wpr, WprH, 4 * NC * NC);

  // 2) gather + ln1 for i, j, c
  ln_in_kernel<<<M, 256, 0, stream>>>(iS, cS, jS, ln1g, ln1b, qsH, kvsH, cLn);

  // 3) projections (Q scaled by 1/sqrt(hd))
  gemm_wmma_kernel<0><<<dim3(M / 128, NC / 64), 128, 0, stream>>>(
      qsH, WqH, bq, qH, nullptr, nullptr, M, NC, NC, 0.125f);
  gemm_wmma_kernel<0><<<dim3(M / 128, (2 * NC) / 64), 128, 0, stream>>>(
      kvsH, WkvH, bkv, kvH, nullptr, nullptr, M, 2 * NC, NC, 1.0f);

  // 4) flash block-causal attention (1 wave per 16-row query tile)
  attn_kernel<<<(NB * NHD * (NT / 16)) / 4, 128, 0, stream>>>(qH, kvH, yH);

  // 5) output projection + residual with ln1(c)
  gemm_wmma_kernel<1><<<dim3(M / 128, NC / 64), 128, 0, stream>>>(
      yH, WoH, bo, nullptr, cs, cLn, M, NC, NC, 1.0f);

  // 6) ln2
  ln_row_kernel<<<M, 256, 0, stream>>>(cs, ln2g, ln2b, cs2H);

  // 7) MLP up + GELU
  gemm_wmma_kernel<2><<<dim3(M / 128, (4 * NC) / 64), 128, 0, stream>>>(
      cs2H, WfcH, bfc, h1H, nullptr, nullptr, M, 4 * NC, NC, 1.0f);

  // 8) MLP down + residual + scatter back to (V,B,C,H,W)
  gemm_wmma_kernel<3><<<dim3(M / 128, NC / 64), 128, 0, stream>>>(
      h1H, WprH, bpr, nullptr, out, cs, M, NC, 4 * NC, 1.0f);
}
